// LUTLayerBasic_59072980189511
// MI455X (gfx1250) — 8-bit matrix operand order repaired and checked by reading, compiled, not run on hardware
//
#include <hip/hip_runtime.h>
#include <hip/hip_bf16.h>

#define B_SZ     2048
#define N_IN     1024
#define N_DET    512
#define N_ANCH   8
#define N_OUT    256
#define N_CHAN   256

typedef int v8i __attribute__((ext_vector_type(8)));

__global__ void lut_bits_kernel(const float* __restrict__ x,
                                unsigned int* __restrict__ bits_u32) {
    int i = blockIdx.x * blockDim.x + threadIdx.x;
    const float4 v = ((const float4*)x)[i];
    unsigned int p = (v.x > 0.0f ? 1u : 0u)
                   | ((v.y > 0.0f ? 1u : 0u) << 8)
                   | ((v.z > 0.0f ? 1u : 0u) << 16)
                   | ((v.w > 0.0f ? 1u : 0u) << 24);
    bits_u32[i] = p;
}

__global__ void lut_smat_kernel(const int* __restrict__ anchors,
                                unsigned char* __restrict__ sT) {
    int d = blockIdx.x * blockDim.x + threadIdx.x;
    if (d >= N_DET) return;
    unsigned char* row = sT + (size_t)d * N_IN;
    uint4 z = {0u, 0u, 0u, 0u};
    for (int i = 0; i < N_IN; i += 16) *(uint4*)(row + i) = z;
    #pragma unroll
    for (int a = 0; a < N_ANCH; ++a) {
        int ia = anchors[d * N_ANCH + a];
        row[ia] = (unsigned char)(row[ia] + (1u << a));
    }
}

__global__ void lut_idx_wmma_kernel(const unsigned char* __restrict__ bits,
                                    const unsigned char* __restrict__ sT,
                                    int* __restrict__ idx) {
    int gtid = blockIdx.x * blockDim.x + threadIdx.x;
    int wave = gtid >> 5;
    int lane = threadIdx.x & 31;
    int bt = wave >> 5;
    int dt = wave & 31;
    int half = (lane >> 4) & 1;
    int m = lane & 15;

    const unsigned char* arow = bits + (size_t)(bt * 16 + m) * N_IN;
    const unsigned char* brow = sT   + (size_t)(dt * 16 + m) * N_IN;

    v8i c;
    #pragma unroll
    for (int r = 0; r < 8; ++r) c[r] = 0;

    for (int k0 = 0; k0 < N_IN; k0 += 64) {
        v8i a, b;
        #pragma unroll
        for (int v = 0; v < 8; ++v) {
            int abase = ((v >> 1) << 4) + ((v & 1) << 2) + half * 8;
            a[v] = *(const int*)(arow + k0 + abase);
            int bbase = ((v >> 1) << 4) + ((v & 1) << 2) + half * 8;
            b[v] = *(const int*)(brow + k0 + bbase);
        }
        c = __builtin_amdgcn_wmma_i32_16x16x64_iu8(false, a, false, b, c,
                                                   false, false);
    }

    #pragma unroll
    for (int r = 0; r < 8; ++r) {
        idx[(size_t)(bt * 16 + r + half * 8) * N_DET + dt * 16 + m] = c[r];
    }
}

__global__ void lut_gather_kernel(const int* __restrict__ idx,
                                  const float* __restrict__ w,
                                  float* __restrict__ out) {
    __shared__ int sidx[16 * N_DET];
    const int t  = threadIdx.x;
    const int bt = blockIdx.x >> 1;
    const int oh = blockIdx.x & 1;
    const int b0 = bt * 16;

    for (int i = t; i < 16 * N_DET; i += 256)
        sidx[i] = idx[(size_t)b0 * N_DET + i];
    __syncthreads();

    const int bl = t >> 4;
    const int o  = oh * 128 + (t & 15) * 8;
    const int* myIdx = sidx + bl * N_DET;

    float a0 = 0.f, a1 = 0.f, a2 = 0.f, a3 = 0.f;
    float a4 = 0.f, a5 = 0.f, a6 = 0.f, a7 = 0.f;

    for (int d = 0; d < N_DET; d += 2) {
        int i0 = myIdx[d];
        int i1 = myIdx[d + 1];
        const float* r0 = w + ((size_t)d * N_CHAN + i0) * N_OUT + o;
        const float* r1 = w + ((size_t)(d + 1) * N_CHAN + i1) * N_OUT + o;
        if (d + 3 < N_DET) {
            int p0 = myIdx[d + 2], p1 = myIdx[d + 3];
            __builtin_prefetch(w + ((size_t)(d + 2) * N_CHAN + p0) * N_OUT + o, 0, 0);
            __builtin_prefetch(w + ((size_t)(d + 3) * N_CHAN + p1) * N_OUT + o, 0, 0);
        }
        float4 u0 = *(const float4*)(r0);
        float4 u1 = *(const float4*)(r0 + 4);
        float4 v0 = *(const float4*)(r1);
        float4 v1 = *(const float4*)(r1 + 4);
        a0 += u0.x + v0.x;  a1 += u0.y + v0.y;
        a2 += u0.z + v0.z;  a3 += u0.w + v0.w;
        a4 += u1.x + v1.x;  a5 += u1.y + v1.y;
        a6 += u1.z + v1.z;  a7 += u1.w + v1.w;
    }

    float* op = out + (size_t)(b0 + bl) * N_OUT + o;
    float4 s0 = {a0, a1, a2, a3};
    float4 s1 = {a4, a5, a6, a7};
    *(float4*)(op)     = s0;
    *(float4*)(op + 4) = s1;
}

extern "C" void kernel_launch(void* const* d_in, const int* in_sizes, int n_in,
                              void* d_out, int out_size, void* d_ws, size_t ws_size,
                              hipStream_t stream) {
    const float* x       = (const float*)d_in[0];
    const int*   anchors = (const int*)d_in[1];
    const float* weights = (const float*)d_in[2];
    float*       out     = (float*)d_out;

    unsigned char* bits = (unsigned char*)d_ws;
    unsigned char* sT   = bits + (size_t)B_SZ * N_IN;
    int*           idx  = (int*)(sT + (size_t)N_DET * N_IN);

    lut_bits_kernel<<<(B_SZ * N_IN / 4) / 256, 256, 0, stream>>>(
        x, (unsigned int*)bits);

    lut_smat_kernel<<<(N_DET + 255) / 256, 256, 0, stream>>>(anchors, sT);

    lut_idx_wmma_kernel<<<512, 256, 0, stream>>>(bits, sT, idx);

    lut_gather_kernel<<<256, 256, 0, stream>>>(idx, weights, out);
}
